// SingleHeadAttention_serialized_37830071943574
// MI455X (gfx1250) — matrix operand order repaired and checked by reading, not compiled, not run on hardware
//
#include <hip/hip_runtime.h>
#include <hip/hip_bf16.h>

// Problem constants (match reference)
#define Bq   64
#define Dq   256
#define Lq   4096
#define DKq  128
#define LN_EPS 1e-6f
#define EPSc   1e-10f

typedef __attribute__((ext_vector_type(16))) _Float16 v16h;
typedef __attribute__((ext_vector_type(8)))  float    v8f;

// ---------------------------------------------------------------------------
// K1: per-(b,l) LayerNorm statistics over D.  Coalesced along L.
// ---------------------------------------------------------------------------
__global__ __launch_bounds__(256) void ln_stats_kernel(
    const float* __restrict__ v, float* __restrict__ mu, float* __restrict__ rsig) {
  int b = blockIdx.x / (Lq / 256);
  int l = (blockIdx.x % (Lq / 256)) * 256 + threadIdx.x;
  const float* vb = v + (size_t)b * Dq * Lq;
  float s = 0.f, s2 = 0.f;
#pragma unroll 4
  for (int d = 0; d < Dq; ++d) {
    float x = vb[(size_t)d * Lq + l];
    s += x; s2 += x * x;
  }
  float m   = s / (float)Dq;
  float var = s2 / (float)Dq - m * m;
  mu[b * Lq + l]   = m;
  rsig[b * Lq + l] = rsqrtf(var + LN_EPS);
}

// ---------------------------------------------------------------------------
// K2: per-(b,d) mean/std of xn over L (for the query's mean_std vector).
// One block per (b,d); 256 threads stream L coalesced; LDS tree reduce.
// ---------------------------------------------------------------------------
__global__ __launch_bounds__(256) void col_stats_kernel(
    const float* __restrict__ v, const float* __restrict__ mu, const float* __restrict__ rsig,
    const float* __restrict__ g, const float* __restrict__ beta,
    float* __restrict__ mean_out, float* __restrict__ std_out) {
  int b = blockIdx.x / Dq;
  int d = blockIdx.x % Dq;
  const float* row = v    + ((size_t)b * Dq + d) * Lq;
  const float* mub = mu   + (size_t)b * Lq;
  const float* rb  = rsig + (size_t)b * Lq;
  float gg = g[d], bb = beta[d];
  float s = 0.f, s2 = 0.f;
  for (int l = threadIdx.x; l < Lq; l += 256) {
    float xn = (row[l] - mub[l]) * rb[l] * gg + bb;
    s += xn; s2 += xn * xn;
  }
  __shared__ float sh0[256], sh1[256];
  int tid = threadIdx.x;
  sh0[tid] = s; sh1[tid] = s2;
  __syncthreads();
  for (int off = 128; off > 0; off >>= 1) {
    if (tid < off) { sh0[tid] += sh0[tid + off]; sh1[tid] += sh1[tid + off]; }
    __syncthreads();
  }
  if (tid == 0) {
    float mean = sh0[0] / (float)Lq;
    float var  = sh1[0] / (float)Lq - mean * mean;
    mean_out[b * Dq + d] = mean;
    std_out [b * Dq + d] = sqrtf(fmaxf(var, EPSc));
  }
}

// ---------------------------------------------------------------------------
// K3: k = xn @ wk^T via v_wmma_f32_16x16x32_f16.
// Block = 256 threads (8 waves), owns 64 l-rows of xn staged in LDS (f16,
// stride 260 halfwords = 130 dwords -> conflict-free bank mapping).
// Wave w owns output columns [16w,16w+16); its wk slice lives in registers
// as pre-packed B fragments (8 K-chunks x 8 dwords).
// Fragment layouts per CDNA5 ISA §7.12.2 (16-bit A 16x32, B 32x16).
// ---------------------------------------------------------------------------
#define LT   64
#define XSTR 260

__global__ __launch_bounds__(256) void gemm_k_kernel(
    const float* __restrict__ v, const float* __restrict__ mu, const float* __restrict__ rsig,
    const float* __restrict__ g, const float* __restrict__ beta,
    const float* __restrict__ wk, float* __restrict__ kout) {
  __shared__ _Float16 xn[LT * XSTR];  // 33,280 B -> many blocks per 320KB WGP
  int b  = blockIdx.x / (Lq / LT);
  int l0 = (blockIdx.x % (Lq / LT)) * LT;
  int tid = threadIdx.x;

  // Phase 1: LayerNorm tile -> LDS f16.  Consecutive threads read consecutive l.
  const float* vb  = v    + (size_t)b * Dq * Lq + l0;
  const float* mub = mu   + (size_t)b * Lq + l0;
  const float* rb  = rsig + (size_t)b * Lq + l0;
  for (int i = tid; i < Dq * LT; i += 256) {
    int d  = i >> 6;         // i / LT
    int ls = i & (LT - 1);   // i % LT
    float x = vb[(size_t)d * Lq + ls];
    float h = (x - mub[ls]) * rb[ls] * g[d] + beta[d];
    xn[ls * XSTR + d] = (_Float16)h;
  }

  // Phase 2: wave-private B fragments from wk (fp32 global -> packed f16 pairs).
  int wave = tid >> 5, lane = tid & 31;
  int halfsel = lane >> 4;   // 0: lanes 0-15, 1: lanes 16-31
  int n = lane & 15;
  unsigned bfr[8][8];
  const float* wkrow = wk + (size_t)(16 * wave + n) * Dq;
#pragma unroll
  for (int c = 0; c < 8; ++c) {
    int dbase = 32 * c + 8 * halfsel;
#pragma unroll
    for (int j = 0; j < 8; ++j) {
      union { _Float16 h[2]; unsigned u; } p;
      int kb = (j < 4 ? 2 * j : 16 + 2 * (j - 4));
      p.h[0] = (_Float16)wkrow[dbase + kb];
      p.h[1] = (_Float16)wkrow[dbase + kb + 1];
      bfr[c][j] = p.u;
    }
  }
  __syncthreads();

  const unsigned* xnu = (const unsigned*)xn;
  int m = lane & 15;
#pragma unroll
  for (int t = 0; t < LT / 16; ++t) {
    v8f acc = {0.f, 0.f, 0.f, 0.f, 0.f, 0.f, 0.f, 0.f};
    int rowdw = ((16 * t + m) * XSTR) >> 1;  // XSTR even -> exact
#pragma unroll
    for (int c = 0; c < 8; ++c) {
      union { unsigned u[8]; v16h vec; } A, Bf;
#pragma unroll
      for (int j = 0; j < 8; ++j) {
        int k = (j < 4 ? 2 * j : 16 + 2 * (j - 4)) + 8 * halfsel;  // A 16x32 f16 layout
        A.u[j]  = xnu[rowdw + ((32 * c + k) >> 1)];
        Bf.u[j] = bfr[c][j];
      }
      acc = __builtin_amdgcn_wmma_f32_16x16x32_f16(
          false, A.vec, false, Bf.vec, (short)0, acc, false, false);
    }
    // C/D layout: VGPR r -> lanes 0-15: (M=r, N=lane); lanes 16-31: (M=8+r, N=lane-16)
    float* kt = kout + ((size_t)b * Lq + l0 + 16 * t) * DKq + 16 * wave;
    int Mbase = 8 * halfsel;
#pragma unroll
    for (int r = 0; r < 8; ++r)
      kt[(size_t)(Mbase + r) * DKq + n] = acc[r];
  }
}

// ---------------------------------------------------------------------------
// K4: q[b,:] = wq @ concat(mean, std).  Tiny.
// ---------------------------------------------------------------------------
__global__ __launch_bounds__(128) void q_kernel(
    const float* __restrict__ wq, const float* __restrict__ mean_, const float* __restrict__ std_,
    float* __restrict__ q) {
  int b = blockIdx.x, kk = threadIdx.x;
  const float* wr = wq + (size_t)kk * (2 * Dq);
  float s = 0.f;
  for (int d = 0; d < Dq; ++d) s += wr[d] * mean_[b * Dq + d];
  for (int d = 0; d < Dq; ++d) s += wr[Dq + d] * std_[b * Dq + d];
  q[b * DKq + kk] = s;
}

// ---------------------------------------------------------------------------
// K5: scores = (k . q)/sqrt(DK); softmax over L.  One block per batch.
// ---------------------------------------------------------------------------
__global__ __launch_bounds__(256) void scores_softmax_kernel(
    const float* __restrict__ kbuf, const float* __restrict__ q,
    float* __restrict__ scores, float* __restrict__ weights) {
  int b = blockIdx.x, tid = threadIdx.x;
  __shared__ float qs[DKq];
  __shared__ float red[256];
  __shared__ float smax, ssum;
  for (int i = tid; i < DKq; i += 256) qs[i] = q[b * DKq + i];
  __syncthreads();
  const float scale = 0.088388347648318447f;  // rsqrt(128)
  float lmax = -3.0e38f;
  for (int l = tid; l < Lq; l += 256) {
    const float* kr = kbuf + ((size_t)b * Lq + l) * DKq;
    float s = 0.f;
#pragma unroll 8
    for (int kk = 0; kk < DKq; ++kk) s += kr[kk] * qs[kk];
    s *= scale;
    scores[b * Lq + l] = s;
    lmax = fmaxf(lmax, s);
  }
  red[tid] = lmax; __syncthreads();
  for (int off = 128; off > 0; off >>= 1) {
    if (tid < off) red[tid] = fmaxf(red[tid], red[tid + off]);
    __syncthreads();
  }
  if (tid == 0) smax = red[0];
  __syncthreads();
  float lsum = 0.f;
  for (int l = tid; l < Lq; l += 256) {
    float e = __expf(scores[b * Lq + l] - smax);
    weights[b * Lq + l] = e;
    lsum += e;
  }
  red[tid] = lsum; __syncthreads();
  for (int off = 128; off > 0; off >>= 1) {
    if (tid < off) red[tid] += red[tid + off];
    __syncthreads();
  }
  if (tid == 0) ssum = red[0];
  __syncthreads();
  float inv = 1.f / ssum;
  for (int l = tid; l < Lq; l += 256) weights[b * Lq + l] *= inv;
}

// ---------------------------------------------------------------------------
// K6: weighted pooling: att_mean/att_std per (b,d).  var = E_w[x^2]-mean^2.
// ---------------------------------------------------------------------------
__global__ __launch_bounds__(256) void att_stats_kernel(
    const float* __restrict__ v, const float* __restrict__ mu, const float* __restrict__ rsig,
    const float* __restrict__ g, const float* __restrict__ beta,
    const float* __restrict__ weights,
    float* __restrict__ att_mean, float* __restrict__ att_std) {
  int b = blockIdx.x / Dq;
  int d = blockIdx.x % Dq;
  const float* row = v    + ((size_t)b * Dq + d) * Lq;
  const float* mub = mu   + (size_t)b * Lq;
  const float* rb  = rsig + (size_t)b * Lq;
  const float* wb  = weights + (size_t)b * Lq;
  float gg = g[d], bb = beta[d];
  float s1 = 0.f, s2 = 0.f;
  for (int l = threadIdx.x; l < Lq; l += 256) {
    float xn = (row[l] - mub[l]) * rb[l] * gg + bb;
    float w  = wb[l];
    s1 += w * xn; s2 += w * xn * xn;
  }
  __shared__ float sh0[256], sh1[256];
  int tid = threadIdx.x;
  sh0[tid] = s1; sh1[tid] = s2;
  __syncthreads();
  for (int off = 128; off > 0; off >>= 1) {
    if (tid < off) { sh0[tid] += sh0[tid + off]; sh1[tid] += sh1[tid + off]; }
    __syncthreads();
  }
  if (tid == 0) {
    float mean = sh0[0];                 // sum(w) == 1
    float var  = sh1[0] - mean * mean;
    att_mean[b * Dq + d] = mean;
    att_std [b * Dq + d] = sqrtf(fmaxf(var, EPSc));
  }
}

// ---------------------------------------------------------------------------
// K7: skip_conn = fc_w @ [att_mean; att_std] + fc_b ; delta = fcq_w @ att_mean
// ---------------------------------------------------------------------------
__global__ __launch_bounds__(256) void heads_kernel(
    const float* __restrict__ fc_w, const float* __restrict__ fc_b,
    const float* __restrict__ fcq_w,
    const float* __restrict__ att_mean, const float* __restrict__ att_std,
    float* __restrict__ out_skip, float* __restrict__ delta) {
  int b = blockIdx.x, o = threadIdx.x;
  const float* fr = fc_w  + (size_t)o * (2 * Dq);
  const float* qr = fcq_w + (size_t)o * Dq;
  float s = fc_b[o], t = 0.f;
  for (int d = 0; d < Dq; ++d) {
    float am = att_mean[b * Dq + d];
    s += fr[d] * am + fr[Dq + d] * att_std[b * Dq + d];
    t += qr[d] * am;
  }
  out_skip[b * Dq + o] = s;
  delta   [b * Dq + o] = t;
}

// ---------------------------------------------------------------------------
// K8: q_out[b,l,o] = v[b,o,l] + delta[b,o], written [B,L,D] via 32x32 LDS tile.
// ---------------------------------------------------------------------------
__global__ __launch_bounds__(256) void out_kernel(
    const float* __restrict__ v, const float* __restrict__ delta, float* __restrict__ out) {
  __shared__ float tile[32][33];
  int b  = blockIdx.z;
  int o0 = blockIdx.y * 32;
  int l0 = blockIdx.x * 32;
  int tx = threadIdx.x & 31, ty = threadIdx.x >> 5;  // 8 rows per pass
#pragma unroll
  for (int r = 0; r < 32; r += 8)
    tile[ty + r][tx] = v[((size_t)b * Dq + o0 + ty + r) * Lq + l0 + tx];
  __syncthreads();
#pragma unroll
  for (int r = 0; r < 32; r += 8) {
    int l = l0 + ty + r;
    int o = o0 + tx;
    out[((size_t)b * Lq + l) * Dq + o] = tile[tx][ty + r] + delta[b * Dq + o];
  }
}

// ---------------------------------------------------------------------------
extern "C" void kernel_launch(void* const* d_in, const int* in_sizes, int n_in,
                              void* d_out, int out_size, void* d_ws, size_t ws_size,
                              hipStream_t stream) {
  (void)in_sizes; (void)n_in; (void)out_size; (void)ws_size;
  const float* v     = (const float*)d_in[0];
  const float* ln_g  = (const float*)d_in[1];
  const float* ln_b  = (const float*)d_in[2];
  const float* wk    = (const float*)d_in[3];
  const float* wq    = (const float*)d_in[4];
  const float* fcq_w = (const float*)d_in[5];
  const float* fc_w  = (const float*)d_in[6];
  const float* fc_b  = (const float*)d_in[7];
  float* out = (float*)d_out;

  // Workspace layout (floats): ~137.6 MB total
  float* ws = (float*)d_ws;
  size_t off = 0;
  float* kbuf     = ws + off; off += (size_t)Bq * Lq * DKq;  // 33.5M
  float* mu       = ws + off; off += (size_t)Bq * Lq;
  float* rsig     = ws + off; off += (size_t)Bq * Lq;
  float* mean_    = ws + off; off += (size_t)Bq * Dq;
  float* std_     = ws + off; off += (size_t)Bq * Dq;
  float* q        = ws + off; off += (size_t)Bq * DKq;
  float* scores   = ws + off; off += (size_t)Bq * Lq;
  float* weights  = ws + off; off += (size_t)Bq * Lq;
  float* att_mean = ws + off; off += (size_t)Bq * Dq;
  float* att_std  = ws + off; off += (size_t)Bq * Dq;
  float* delta    = ws + off; off += (size_t)Bq * Dq;

  float* out_qout = out;                          // [B, L, D]
  float* out_skip = out + (size_t)Bq * Lq * Dq;   // [B, D, 1]

  ln_stats_kernel<<<Bq * (Lq / 256), 256, 0, stream>>>(v, mu, rsig);
  col_stats_kernel<<<Bq * Dq, 256, 0, stream>>>(v, mu, rsig, ln_g, ln_b, mean_, std_);
  gemm_k_kernel<<<Bq * (Lq / LT), 256, 0, stream>>>(v, mu, rsig, ln_g, ln_b, wk, kbuf);
  q_kernel<<<Bq, 128, 0, stream>>>(wq, mean_, std_, q);
  scores_softmax_kernel<<<Bq, 256, 0, stream>>>(kbuf, q, scores, weights);
  att_stats_kernel<<<Bq * Dq, 256, 0, stream>>>(v, mu, rsig, ln_g, ln_b, weights,
                                                att_mean, att_std);
  heads_kernel<<<Bq, 256, 0, stream>>>(fc_w, fc_b, fcq_w, att_mean, att_std,
                                       out_skip, delta);
  out_kernel<<<dim3(Lq / 32, Dq / 32, Bq), 256, 0, stream>>>(v, delta, out_qout);
}
